// EventImageFusion_23381801959728
// MI455X (gfx1250) — compile-verified, matrix operand order checked by reading, not run on hardware
//
#include <hip/hip_runtime.h>
#include <hip/hip_bf16.h>
#include <math.h>

typedef __attribute__((ext_vector_type(16))) _Float16 v16h;
typedef __attribute__((ext_vector_type(8)))  float    v8f;

// ---------------------------------------------------------------------------
// WMMA helpers (gfx1250, wave32).  16x16x32 f16 -> f32 accumulate.
// A/B per-lane layout (ISA 7.12.2): lane = hi*16 + lm ; the 16 halves cover
// K = hi*8 + {0..7, 16..23} (dword-contiguous pairs in a K-contiguous row ->
// the compiler merges these into 2x ds_load_b128 per fragment).
// C/D: lane holds col = lm, rows r + hi*8 in regs 0..7.
// ---------------------------------------------------------------------------
__device__ __forceinline__ v16h frag_load(const _Float16* rowPtr, int hi) {
  const unsigned int* p = (const unsigned int*)rowPtr;  // 32 contiguous halves
  union { v16h v; unsigned int u[8]; } r;
  const int b = hi << 2;
#pragma unroll
  for (int i = 0; i < 4; ++i) r.u[i] = p[b + i];
#pragma unroll
  for (int i = 0; i < 4; ++i) r.u[4 + i] = p[b + 8 + i];
  return r.v;
}

__device__ __forceinline__ v8f wmma_f16(v16h a, v16h b, v8f c) {
  return __builtin_amdgcn_wmma_f32_16x16x32_f16(false, a, false, b, (short)0, c,
                                                false, false);
}

// ---------------------------------------------------------------------------
// CDNA5 async memory->LDS staging (ASYNCcnt path).  Each lane moves one 16B
// chunk per op; weights are pre-packed so a whole LDS tile is one contiguous
// global slab.  LDS dest operand = low 32 bits of the flat shared pointer
// (ISA: LDS flat addresses carry the byte offset in addr[31:0]).
// ---------------------------------------------------------------------------
__device__ __forceinline__ void async_tile16(const void* gsrc, void* ldst,
                                             int bytes, int tid) {
  const unsigned la0 = (unsigned)(size_t)ldst;
  const unsigned long long ga0 = (unsigned long long)(size_t)gsrc;
  for (int off = tid * 16; off < bytes; off += 256 * 16) {
    unsigned la = la0 + (unsigned)off;
    unsigned long long ga = ga0 + (unsigned long long)(unsigned)off;
    asm volatile("global_load_async_to_lds_b128 %0, %1, off"
                 :: "v"(la), "v"(ga) : "memory");
  }
}
__device__ __forceinline__ void wait_async0() {
  asm volatile("s_wait_asynccnt 0x0" ::: "memory");
}

__device__ __forceinline__ int regid(int c) {  // shift-mask region, w=8 shift=4
  return (c < 120) ? 0 : (c < 124 ? 1 : 2);
}

// ---------------------------------------------------------------------------
// Weight packing: f32 [K][N] (n fastest) -> f16 slabs [K/32][N][32]
// so one GEMM K-step tile == one contiguous async-copyable slab.
// ---------------------------------------------------------------------------
__global__ void pack_w_f16(const float* __restrict__ W, _Float16* __restrict__ O,
                           int K, int N) {
  const int total = K * N;
  for (int i = blockIdx.x * blockDim.x + threadIdx.x; i < total;
       i += gridDim.x * blockDim.x) {
    const int kb = i / (N * 32);
    const int r = i % (N * 32);
    const int n = r >> 5, kk = r & 31;
    O[i] = (_Float16)W[(kb * 32 + kk) * N + n];
  }
}

// conv weights (128,512,3,3) -> [144][128][32] with kb = tap*16 + cinblk
__global__ void pack_convw_f16(const float* __restrict__ W,
                               _Float16* __restrict__ O) {
  const int total = 144 * 128 * 32;
  for (int i = blockIdx.x * blockDim.x + threadIdx.x; i < total;
       i += gridDim.x * blockDim.x) {
    const int kb = i >> 12, r = i & 4095;
    const int n = r >> 5, kk = r & 31;
    const int tap = kb >> 4, cb = kb & 15;
    O[i] = (_Float16)W[(n * 512 + cb * 32 + kk) * 9 + tap];
  }
}

// ---------------------------------------------------------------------------
// Kernel 1: 3x3 conv (Cin=512 -> C=128) as implicit GEMM, tokens (B,L,C) f32.
// One block = one image row.  Weight slabs double-buffered via async->LDS;
// im2col tile double-buffered via ds_store; ONE barrier per K-step.
// ---------------------------------------------------------------------------
__global__ __launch_bounds__(256)
void conv3x3_tokens(const float* __restrict__ X, const _Float16* __restrict__ Wc,
                    const float* __restrict__ Bv, float* __restrict__ Y) {
  __shared__ _Float16 al[2][4096];  // A tile: [pixel][k]
  __shared__ _Float16 bl[2][4096];  // W tile: [n][k]
  const int tid = threadIdx.x;
  const int wv = tid >> 5, lane = tid & 31, hi = lane >> 4, lm = lane & 15;
  const int t = blockIdx.x;
  const int b = t >> 7, h = t & 127;

  async_tile16(Wc, &bl[0][0], 8192, tid);  // prologue: slab 0 in flight

  v8f acc[8] = {};
  for (int kb = 0; kb < 144; ++kb) {
    const int cur = kb & 1;
    const int tap = kb >> 4, cin0 = (kb & 15) << 5;
    const int dh = tap / 3, dw = tap % 3;
    const int hh = h + dh - 1;
    _Float16* alc = &al[cur][0];
    for (int i = tid; i < 4096; i += 256) {  // im2col + f16 cvt (overlaps async)
      const int p = i >> 5, kk = i & 31;
      const int ww = p + dw - 1;
      float v = 0.f;
      if (hh >= 0 && hh < 128 && (unsigned)ww < 128u)
        v = X[(((b * 512 + cin0 + kk) * 128 + hh) << 7) + ww];
      alc[i] = (_Float16)v;
    }
    wait_async0();     // this wave's async copies of slab kb done
    __syncthreads();   // all waves: tile kb (A + W) visible, prev reads done
    if (kb + 1 < 144)
      async_tile16(Wc + (kb + 1) * 4096, &bl[1 ^ cur][0], 8192, tid);
    v16h a = frag_load(&alc[(wv * 16 + lm) * 32], hi);
#pragma unroll
    for (int nt = 0; nt < 8; ++nt) {
      v16h bf = frag_load(&bl[cur][(nt * 16 + lm) * 32], hi);
      acc[nt] = wmma_f16(a, bf, acc[nt]);
    }
  }
#pragma unroll
  for (int nt = 0; nt < 8; ++nt)
#pragma unroll
    for (int r = 0; r < 8; ++r) {
      const int p = wv * 16 + r + hi * 8;
      const int n = nt * 16 + lm;
      Y[((b * 16384 + h * 128 + p) << 7) + n] = acc[nt][r] + Bv[n];
    }
}

// ---------------------------------------------------------------------------
// Kernel 2: one LeWin attention block per 8x8 window.  64 KB LDS arena.
// ---------------------------------------------------------------------------
__global__ __launch_bounds__(256)
void lewin_attn(const float* __restrict__ X, const float* __restrict__ ln_g,
                const float* __restrict__ ln_b, const _Float16* __restrict__ Wq,
                const float* __restrict__ qkv_b, const _Float16* __restrict__ Wp,
                const float* __restrict__ proj_b, const float* __restrict__ rpb,
                float* __restrict__ Y, int shift) {
  __shared__ __align__(16) char smem[65536];
  _Float16* xln = (_Float16*)smem;            // [64][128]   phase 1-2
  _Float16* wtb0 = (_Float16*)(smem + 16384); // [384][32]   phase 2 buf0
  _Float16* wtb1 = (_Float16*)(smem + 40960); // [384][32]   phase 2 buf1
  _Float16* qb  = (_Float16*)smem;            // [4][64][32] phase 3-4
  _Float16* kbf = (_Float16*)(smem + 16384);  // [4][64][32] phase 3-4
  _Float16* vT  = (_Float16*)(smem + 32768);  // [4][32][64] phase 3-5
  _Float16* pb  = (_Float16*)smem;            // [4][64][64] phase 4-5
  _Float16* xat = (_Float16*)(smem + 32768);  // [64][128]   phase 5-6
  _Float16* pwt0 = (_Float16*)(smem + 49152); // [128][32]   phase 6 buf0
  _Float16* pwt1 = (_Float16*)(smem + 57344); // [128][32]   phase 6 buf1
  _Float16* wtb[2] = {wtb0, wtb1};
  _Float16* pwtb[2] = {pwt0, pwt1};

  const int tid = threadIdx.x;
  const int wv = tid >> 5, lane = tid & 31, hi = lane >> 4, lm = lane & 15;
  const int wid = blockIdx.x;
  const int b = wid >> 8, wy = (wid >> 4) & 15, wx = wid & 15;

  async_tile16(Wq, wtb0, 24576, tid);  // QKV slab 0 in flight during LN

  // ---- Phase 1: LayerNorm (4 lanes per token, shfl reduce) ----
  {
    const int tok = tid >> 2, part = tid & 3;
    const int iy = tok >> 3, ix = tok & 7;
    const int hh = (wy * 8 + iy + shift) & 127;
    const int ww = (wx * 8 + ix + shift) & 127;
    const float* xrow = X + ((b * 16384 + hh * 128 + ww) << 7);
    float s = 0.f, s2 = 0.f;
    for (int c = part * 32; c < part * 32 + 32; ++c) {
      float v = xrow[c]; s += v; s2 += v * v;
    }
    s  += __shfl_xor(s, 1);  s  += __shfl_xor(s, 2);
    s2 += __shfl_xor(s2, 1); s2 += __shfl_xor(s2, 2);
    const float mean = s * (1.f / 128.f);
    const float inv = rsqrtf(s2 * (1.f / 128.f) - mean * mean + 1e-5f);
    for (int c = part * 32; c < part * 32 + 32; ++c)
      xln[tok * 128 + c] = (_Float16)((xrow[c] - mean) * inv * ln_g[c] + ln_b[c]);
  }

  // ---- Phase 2: QKV GEMM 64x128 @ 128x384, double-buffered weights ----
  v8f acc[12] = {};
  const int mt = wv & 3, ng = wv >> 2;
  for (int ks = 0; ks < 4; ++ks) {
    wait_async0();
    __syncthreads();  // slab ks landed (all waves); prev-tile reads done
    if (ks < 3) async_tile16(Wq + (ks + 1) * 12288, wtb[(ks + 1) & 1], 24576, tid);
    const int k0 = ks << 5;
    v16h a = frag_load(&xln[(mt * 16 + lm) * 128 + k0], hi);
    const _Float16* wt = wtb[ks & 1];
#pragma unroll
    for (int j = 0; j < 12; ++j) {
      v16h bf = frag_load(&wt[((ng * 12 + j) * 16 + lm) * 32], hi);
      acc[j] = wmma_f16(a, bf, acc[j]);
    }
  }
  __syncthreads();  // everyone done reading xln / weight bufs

  // ---- Phase 3: scatter q (scaled), k, v^T into LDS ----
#pragma unroll
  for (int j = 0; j < 12; ++j)
#pragma unroll
    for (int r = 0; r < 8; ++r) {
      const int m = mt * 16 + r + hi * 8;
      const int n = (ng * 12 + j) * 16 + lm;
      const float v = acc[j][r] + qkv_b[n];
      const int which = n >> 7, hn = n & 127, head = hn >> 5, d = hn & 31;
      if (which == 0)      qb[(head * 64 + m) * 32 + d] = (_Float16)(v * 0.17677669529663687f);
      else if (which == 1) kbf[(head * 64 + m) * 32 + d] = (_Float16)v;
      else                 vT[(head * 32 + d) * 64 + m] = (_Float16)v;
    }
  __syncthreads();

  // ---- Phase 4: scores = q @ k^T (K=32 -> single WMMA per tile) ----
  const int head = wv >> 1, mh = wv & 1;  // 2 waves per head, 32 rows each
  v8f sc[2][4] = {};
#pragma unroll
  for (int m2 = 0; m2 < 2; ++m2) {
    v16h a = frag_load(&qb[(head * 64 + (mh * 2 + m2) * 16 + lm) * 32], hi);
#pragma unroll
    for (int nt = 0; nt < 4; ++nt) {
      v16h bf = frag_load(&kbf[(head * 64 + nt * 16 + lm) * 32], hi);
      sc[m2][nt] = wmma_f16(a, bf, sc[m2][nt]);
    }
  }
  __syncthreads();  // q/k dead -> prob region reusable

  // bias + mask + register softmax (row spans 16 lanes x 4 tiles)
#pragma unroll
  for (int m2 = 0; m2 < 2; ++m2)
#pragma unroll
    for (int r = 0; r < 8; ++r) {
      const int m = (mh * 2 + m2) * 16 + r + hi * 8;
      const int iy = m >> 3, ix = m & 7;
      int cnti = 0;
      if (shift > 0) cnti = regid(wy * 8 + iy) * 3 + regid(wx * 8 + ix);
      float v[4];
#pragma unroll
      for (int nt = 0; nt < 4; ++nt) {
        const int col = nt * 16 + lm;
        const int jy = col >> 3, jx = col & 7;
        const int rel = (iy - jy + 7) * 15 + (ix - jx + 7);
        float bias = rpb[rel * 4 + head];
        if (shift > 0) {
          const int cntj = regid(wy * 8 + jy) * 3 + regid(wx * 8 + jx);
          if (cnti != cntj) bias -= 100.f;
        }
        v[nt] = sc[m2][nt][r] + bias;
      }
      float mx = fmaxf(fmaxf(v[0], v[1]), fmaxf(v[2], v[3]));
      mx = fmaxf(mx, __shfl_xor(mx, 1));
      mx = fmaxf(mx, __shfl_xor(mx, 2));
      mx = fmaxf(mx, __shfl_xor(mx, 4));
      mx = fmaxf(mx, __shfl_xor(mx, 8));
      float sum = 0.f;
#pragma unroll
      for (int nt = 0; nt < 4; ++nt) { v[nt] = __expf(v[nt] - mx); sum += v[nt]; }
      sum += __shfl_xor(sum, 1);
      sum += __shfl_xor(sum, 2);
      sum += __shfl_xor(sum, 4);
      sum += __shfl_xor(sum, 8);
      const float rs = 1.f / sum;
#pragma unroll
      for (int nt = 0; nt < 4; ++nt)
        pb[(head * 64 + m) * 64 + nt * 16 + lm] = (_Float16)(v[nt] * rs);
    }
  __syncthreads();

  // ---- Phase 5: out = P @ V (K=64 -> 2 WMMA K-steps) ----
  v8f ov[2][2] = {};
#pragma unroll
  for (int ks = 0; ks < 2; ++ks) {
    const int k0 = ks << 5;
#pragma unroll
    for (int m2 = 0; m2 < 2; ++m2) {
      v16h a = frag_load(&pb[(head * 64 + (mh * 2 + m2) * 16 + lm) * 64 + k0], hi);
#pragma unroll
      for (int dt = 0; dt < 2; ++dt) {
        v16h bf = frag_load(&vT[(head * 32 + dt * 16 + lm) * 64 + k0], hi);
        ov[m2][dt] = wmma_f16(a, bf, ov[m2][dt]);
      }
    }
  }
  __syncthreads();  // v^T dead -> xat region reusable
  async_tile16(Wp, pwt0, 8192, tid);  // proj slab 0 in flight
#pragma unroll
  for (int m2 = 0; m2 < 2; ++m2)
#pragma unroll
    for (int dt = 0; dt < 2; ++dt)
#pragma unroll
      for (int r = 0; r < 8; ++r) {
        const int m = (mh * 2 + m2) * 16 + r + hi * 8;
        const int c = head * 32 + dt * 16 + lm;
        xat[m * 128 + c] = (_Float16)ov[m2][dt][r];
      }

  // ---- Phase 6: proj GEMM 64x128 @ 128x128, double-buffered weights ----
  v8f po[4] = {};
  const int nq = wv >> 2;
  for (int ks = 0; ks < 4; ++ks) {
    wait_async0();
    __syncthreads();  // xat + proj slab ks visible
    if (ks < 3) async_tile16(Wp + (ks + 1) * 4096, pwtb[(ks + 1) & 1], 8192, tid);
    const int k0 = ks << 5;
    v16h a = frag_load(&xat[(mt * 16 + lm) * 128 + k0], hi);
    const _Float16* pw = pwtb[ks & 1];
#pragma unroll
    for (int j = 0; j < 4; ++j) {
      v16h bf = frag_load(&pw[((nq * 4 + j) * 16 + lm) * 32], hi);
      po[j] = wmma_f16(a, bf, po[j]);
    }
  }
#pragma unroll
  for (int j = 0; j < 4; ++j)
#pragma unroll
    for (int r = 0; r < 8; ++r) {
      const int m = mt * 16 + r + hi * 8;
      const int n = (nq * 4 + j) * 16 + lm;
      const int iy = m >> 3, ix = m & 7;
      const int g = (b << 14) + (((wy * 8 + iy + shift) & 127) << 7) +
                    ((wx * 8 + ix + shift) & 127);
      Y[(g << 7) + n] = X[(g << 7) + n] + po[j][r] + proj_b[n];
    }
}

// ---------------------------------------------------------------------------
// Kernel 3: LN2 + MLP (128->512, exact GELU, 512->128) + residual.
// fc1 weights async (single buffer, 4 steps); fc2 async double-buffered (16).
// ---------------------------------------------------------------------------
__global__ __launch_bounds__(256)
void lewin_mlp(const float* __restrict__ X, const float* __restrict__ ln_g,
               const float* __restrict__ ln_b, const _Float16* __restrict__ W1,
               const float* __restrict__ fc1_b, const _Float16* __restrict__ W2,
               const float* __restrict__ fc2_b, float* __restrict__ Y,
               int transposed_out) {
  __shared__ __align__(16) char smem[57344];
  _Float16* xln = (_Float16*)smem;            // [32][128]  8 KB
  float*    xf  = (float*)(smem + 8192);      // [32][128] 16 KB (phase 1)
  _Float16* w1t = (_Float16*)(smem + 24576);  // [512][32] 32 KB (fc1)
  _Float16* hb  = (_Float16*)(smem + 8192);   // [32][512] 32 KB (fc2 A)
  _Float16* w2t0 = (_Float16*)(smem + 40960); // [128][32]  8 KB
  _Float16* w2t1 = (_Float16*)(smem + 49152); // [128][32]  8 KB
  _Float16* w2tb[2] = {w2t0, w2t1};

  const int tid = threadIdx.x;
  const int wv = tid >> 5, lane = tid & 31, hi = lane >> 4, lm = lane & 15;
  const int g0 = blockIdx.x * 32;

  async_tile16(W1, w1t, 32768, tid);  // fc1 slab 0 overlaps LN

  for (int i = tid; i < 32 * 128; i += 256) xf[i] = X[(g0 << 7) + i];
  __syncthreads();
  if (tid < 128) {
    const int tok = tid >> 2, part = tid & 3;
    float s = 0.f, s2 = 0.f;
    for (int c = part * 32; c < part * 32 + 32; ++c) {
      float v = xf[tok * 128 + c]; s += v; s2 += v * v;
    }
    s  += __shfl_xor(s, 1);  s  += __shfl_xor(s, 2);
    s2 += __shfl_xor(s2, 1); s2 += __shfl_xor(s2, 2);
    const float mean = s * (1.f / 128.f);
    const float inv = rsqrtf(s2 * (1.f / 128.f) - mean * mean + 1e-5f);
    for (int c = part * 32; c < part * 32 + 32; ++c)
      xln[tok * 128 + c] =
          (_Float16)((xf[tok * 128 + c] - mean) * inv * ln_g[c] + ln_b[c]);
  }

  // fc1: 32x128 @ 128x512 (single-buffer async weight slabs)
  v8f a1[8] = {};
  const int mt = wv & 1, no = wv >> 1;
  for (int ks = 0; ks < 4; ++ks) {
    wait_async0();
    __syncthreads();  // slab ks + (ks==0: xln) visible
    const int k0 = ks << 5;
    v16h a = frag_load(&xln[(mt * 16 + lm) * 128 + k0], hi);
#pragma unroll
    for (int j = 0; j < 8; ++j) {
      v16h bf = frag_load(&w1t[((no * 8 + j) * 16 + lm) * 32], hi);
      a1[j] = wmma_f16(a, bf, a1[j]);
    }
    __syncthreads();  // all done reading w1t before overwrite
    if (ks < 3) async_tile16(W1 + (ks + 1) * 16384, w1t, 32768, tid);
  }
  async_tile16(W2, w2t0, 8192, tid);  // fc2 slab 0 overlaps GELU

  // exact GELU -> h (f16)
#pragma unroll
  for (int j = 0; j < 8; ++j)
#pragma unroll
    for (int r = 0; r < 8; ++r) {
      const int m = mt * 16 + r + hi * 8;
      const int n = (no * 8 + j) * 16 + lm;
      const float v = a1[j][r] + fc1_b[n];
      hb[m * 512 + n] = (_Float16)(0.5f * v * (1.f + erff(v * 0.70710678118654752f)));
    }

  // fc2: 32x512 @ 512x128 (double-buffered async weight slabs)
  v8f a2[2] = {};
  const int np = wv >> 1;
  for (int ks = 0; ks < 16; ++ks) {
    wait_async0();
    __syncthreads();  // hb + slab ks visible; prev-tile reads done
    if (ks < 15) async_tile16(W2 + (ks + 1) * 4096, w2tb[(ks + 1) & 1], 8192, tid);
    const int k0 = ks << 5;
    v16h a = frag_load(&hb[(mt * 16 + lm) * 512 + k0], hi);
    const _Float16* w2 = w2tb[ks & 1];
#pragma unroll
    for (int j = 0; j < 2; ++j) {
      v16h bf = frag_load(&w2[((np * 2 + j) * 16 + lm) * 32], hi);
      a2[j] = wmma_f16(a, bf, a2[j]);
    }
  }
#pragma unroll
  for (int j = 0; j < 2; ++j)
#pragma unroll
    for (int r = 0; r < 8; ++r) {
      const int m = mt * 16 + r + hi * 8;
      const int n = (np * 2 + j) * 16 + lm;
      const int g = g0 + m;
      const float v = a2[j][r] + fc2_b[n] + X[(g << 7) + n];
      if (transposed_out) {
        const int bI = g >> 14, rem = g & 16383, hh = rem >> 7, ww = rem & 127;
        Y[(((bI * 128 + n) * 128 + hh) << 7) + ww] = v;
      } else {
        Y[(g << 7) + n] = v;
      }
    }
}

// ---------------------------------------------------------------------------
extern "C" void kernel_launch(void* const* d_in, const int* in_sizes, int n_in,
                              void* d_out, int out_size, void* d_ws, size_t ws_size,
                              hipStream_t stream) {
  (void)in_sizes; (void)n_in; (void)out_size; (void)ws_size;
  const float* x      = (const float*)d_in[0];
  const float* conv_w = (const float*)d_in[1];
  const float* conv_b = (const float*)d_in[2];

  const size_t NTOK = (size_t)8 * 128 * 128;
  float* bufA = (float*)d_ws;              // token buffer (B,L,C) f32
  float* bufB = bufA + NTOK * 128;         // second token buffer
  _Float16* wbase = (_Float16*)(bufB + NTOK * 128);
  _Float16* wc = wbase;                    // conv:   144*4096 halves
  // per block: qkv 4*12288, proj 4*4096, fc1 4*16384, fc2 16*4096 = 196608
  _Float16* blkw[2] = {wbase + 589824, wbase + 589824 + 196608};

  // ---- pack all weights to f16 GEMM slabs (tiny, once per launch) ----
  pack_convw_f16<<<1152, 256, 0, stream>>>(conv_w, wc);
  for (int blk = 0; blk < 2; ++blk) {
    const int base = 3 + blk * 13;
    pack_w_f16<<<96, 256, 0, stream>>>((const float*)d_in[base + 2],
                                       blkw[blk], 128, 384);             // qkv
    pack_w_f16<<<32, 256, 0, stream>>>((const float*)d_in[base + 4],
                                       blkw[blk] + 49152, 128, 128);     // proj
    pack_w_f16<<<128, 256, 0, stream>>>((const float*)d_in[base + 9],
                                        blkw[blk] + 65536, 128, 512);    // fc1
    pack_w_f16<<<128, 256, 0, stream>>>((const float*)d_in[base + 11],
                                        blkw[blk] + 131072, 512, 128);   // fc2
  }

  conv3x3_tokens<<<1024, 256, 0, stream>>>(x, wc, conv_b, bufA);

  for (int blk = 0; blk < 2; ++blk) {
    const int base = 3 + blk * 13;
    const float* ln1_g = (const float*)d_in[base + 0];
    const float* ln1_b = (const float*)d_in[base + 1];
    const float* qkv_b = (const float*)d_in[base + 3];
    const float* projb = (const float*)d_in[base + 5];
    const float* rpb   = (const float*)d_in[base + 6];
    const float* ln2_g = (const float*)d_in[base + 7];
    const float* ln2_b = (const float*)d_in[base + 8];
    const float* fc1b  = (const float*)d_in[base + 10];
    const float* fc2b  = (const float*)d_in[base + 12];

    lewin_attn<<<2048, 256, 0, stream>>>(bufA, ln1_g, ln1_b, blkw[blk], qkv_b,
                                         blkw[blk] + 49152, projb, rpb, bufB,
                                         blk == 0 ? 0 : 4);
    if (blk == 0) {
      lewin_mlp<<<4096, 256, 0, stream>>>(bufB, ln2_g, ln2_b, blkw[blk] + 65536,
                                          fc1b, blkw[blk] + 131072, fc2b,
                                          bufA, 0);
    } else {
      lewin_mlp<<<4096, 256, 0, stream>>>(bufB, ln2_g, ln2_b, blkw[blk] + 65536,
                                          fc1b, blkw[blk] + 131072, fc2b,
                                          (float*)d_out, 1);
    }
  }
}
